// DecoderTransformerBlock_26199300506357
// MI455X (gfx1250) — compile-verified, matrix operand order checked by reading, not run on hardware
//
#include <hip/hip_runtime.h>

#define B_   2
#define S_   2048
#define DM_  1024
#define H_   16
#define HD_  64
#define DFF_ 4096

typedef __attribute__((ext_vector_type(16))) __bf16 v16bf;
typedef __attribute__((ext_vector_type(8)))  __bf16 v8bf;
typedef __attribute__((ext_vector_type(8)))  float  v8f;

union Frag { v16bf v; __bf16 h[16]; };

// Load a 16-element bf16 fragment whose two 8-element halves live at
// base[0..7] and base[16..23] (both 16-byte aligned) -> two ds_load_b128.
__device__ __forceinline__ v16bf load_frag_lds(const __bf16* base) {
  const v8bf lo = *(const v8bf*)(base);
  const v8bf hi = *(const v8bf*)(base + 16);
  return __builtin_shufflevector(lo, hi, 0, 1, 2, 3, 4, 5, 6, 7,
                                 8, 9, 10, 11, 12, 13, 14, 15);
}

// Build a fragment from fp32 global memory: halves at p[0..7], p[16..23].
__device__ __forceinline__ v16bf frag_from_f32(const float* __restrict__ p) {
  Frag f;
  #pragma unroll
  for (int e = 0; e < 8; ++e) {
    f.h[e]     = (__bf16)p[e];
    f.h[e + 8] = (__bf16)p[16 + e];
  }
  return f.v;
}

// Within-wave LDS producer->consumer ordering (wave32 lockstep, no barrier needed)
__device__ __forceinline__ void lds_fence() {
  asm volatile("s_wait_dscnt 0" ::: "memory");
}

// ---------------------------------------------------------------------------
// GEMM: C[M,N] = A[M,K] @ W[K,N] + bias (+ReLU).  fp32 in/out, bf16 WMMA core.
// Block: 256 threads (8 waves), tile 128x64; wave tile 32x32 = 2x2 WMMA 16x16.
// A tile row-major in LDS; B tile stored TRANSPOSED (BsT[n][k]) so both A and
// B fragment gathers are contiguous 16B runs -> ds_load_b128 pairs.
// ---------------------------------------------------------------------------
#define BM 128
#define BN 64
#define BK 32
#define AST 40   // 80B row stride (16B-aligned runs)
#define BTST 40

__global__ __launch_bounds__(256) void gemm_bf16_kernel(
    const float* __restrict__ A, const float* __restrict__ W,
    const float* __restrict__ bias, float* __restrict__ C,
    int M, int K, int N, int relu)
{
  __shared__ __align__(16) __bf16 As[BM * AST];
  __shared__ __align__(16) __bf16 BsT[BN * BTST];

  const int tid  = threadIdx.x;
  const int lane = tid & 31, wave = tid >> 5;
  const int wm = wave & 3, wn = wave >> 2;     // 4x2 wave grid
  const int lr = lane & 15, hi = lane >> 4;    // wave32 halves
  const int kb = hi * 8;                       // K base per ISA A/B layout

  const int brow = blockIdx.y * BM;
  const int bcol = blockIdx.x * BN;

  v8f acc[2][2];
  #pragma unroll
  for (int i = 0; i < 2; ++i)
    #pragma unroll
    for (int j = 0; j < 2; ++j)
      #pragma unroll
      for (int e = 0; e < 8; ++e) acc[i][j][e] = 0.f;

  const int arow  = tid >> 1, acol0 = (tid & 1) * 16;   // 2 threads/row, 16 elems
  const int bkrow = tid >> 3, bcol0 = (tid & 7) * 8;    // 8 threads/row, 8 elems

  for (int k0 = 0; k0 < K; k0 += BK) {
    // A tile: coalesced row reads, row-major LDS store (vectorizable)
    const float* ap = A + (size_t)(brow + arow) * K + k0 + acol0;
    __bf16* asr = &As[arow * AST + acol0];
    #pragma unroll
    for (int i = 0; i < 16; ++i) asr[i] = (__bf16)ap[i];

    // B tile: coalesced row reads, transposed LDS store BsT[n][k]
    const float* bp = W + (size_t)(k0 + bkrow) * N + bcol + bcol0;
    #pragma unroll
    for (int i = 0; i < 8; ++i) BsT[(bcol0 + i) * BTST + bkrow] = (__bf16)bp[i];

    if (k0 + BK < K) {                         // CDNA5 global_prefetch path
      __builtin_prefetch(A + (size_t)(brow + arow) * K + k0 + BK + acol0, 0, 3);
      __builtin_prefetch(W + (size_t)(k0 + BK + bkrow) * N + bcol + bcol0, 0, 3);
    }
    __syncthreads();

    v16bf a[2], b[2];
    #pragma unroll
    for (int t = 0; t < 2; ++t)                // A 16x32 fragment: M=lr
      a[t] = load_frag_lds(&As[(wm * 32 + t * 16 + lr) * AST + kb]);
    #pragma unroll
    for (int t = 0; t < 2; ++t)                // B 32x16 fragment: N=lr
      b[t] = load_frag_lds(&BsT[(wn * 32 + t * 16 + lr) * BTST + kb]);

    #pragma unroll
    for (int i = 0; i < 2; ++i)
      #pragma unroll
      for (int j = 0; j < 2; ++j)
        acc[i][j] = __builtin_amdgcn_wmma_f32_16x16x32_bf16(
            false, a[i], false, b[j], (short)0, acc[i][j], false, false);
    __syncthreads();
  }

  #pragma unroll
  for (int i = 0; i < 2; ++i) {
    #pragma unroll
    for (int j = 0; j < 2; ++j) {
      const int cg = bcol + wn * 32 + j * 16 + lr;
      const float bv = bias[cg];
      #pragma unroll
      for (int e = 0; e < 8; ++e) {            // C/D: M = e + 8*hi, N = lr
        const int rg = brow + wm * 32 + i * 16 + hi * 8 + e;
        float o = acc[i][j][e] + bv;
        if (relu) o = o > 0.f ? o : 0.f;
        C[(size_t)rg * N + cg] = o;
      }
    }
  }
}

// ---------------------------------------------------------------------------
// Flash attention: one wave per (b, h, 16-query tile); online softmax over
// 32-key chunks. Reproduces the reference's plain-reshape head split:
// elem(b,h,s,hd) = buf[b*S*DM + h*S*64 + s*64 + hd].
// ---------------------------------------------------------------------------
__global__ __launch_bounds__(128) void attn_kernel(
    const float* __restrict__ q, const float* __restrict__ k,
    const float* __restrict__ v, float* __restrict__ out, int causal)
{
  __shared__ __align__(16) float  Sb[4][16][33];
  __shared__ __align__(16) __bf16 Pb[4][16][40];   // 80B rows: aligned frag runs
  __shared__ float Al[4][16];
  __shared__ float Ll[4][16];

  const int lane = threadIdx.x & 31;
  const int w    = threadIdx.x >> 5;
  const int wid  = blockIdx.x * 4 + w;
  const int qt = wid & 127;                    // S/16 = 128 q-tiles
  const int h  = (wid >> 7) & 15;
  const int b  = wid >> 11;

  const int lr = lane & 15, hi = lane >> 4, kb = hi * 8;

  const float* qh = q + (size_t)b * S_ * DM_ + (size_t)h * S_ * HD_;
  const float* kh = k + (size_t)b * S_ * DM_ + (size_t)h * S_ * HD_;
  const float* vh = v + (size_t)b * S_ * DM_ + (size_t)h * S_ * HD_;

  v16bf qa[2];                                 // Q 16x64 -> two K=32 steps
  #pragma unroll
  for (int s = 0; s < 2; ++s)
    qa[s] = frag_from_f32(qh + (size_t)(qt * 16 + lr) * HD_ + s * 32 + kb);

  v8f o[4];
  #pragma unroll
  for (int n = 0; n < 4; ++n)
    #pragma unroll
    for (int e = 0; e < 8; ++e) o[n][e] = 0.f;
  float m = -1e30f, l = 0.f;
  const int qglob = qt * 16 + lr;

  const int jend = causal ? ((qt * 16 + 15) / 32) : (S_ / 32 - 1);
  for (int j = 0; j <= jend; ++j) {
    // S = Q @ K^T for 32 keys (two 16-key N tiles x two K=32 steps)
    v8f sc[2];
    #pragma unroll
    for (int n = 0; n < 2; ++n)
      #pragma unroll
      for (int e = 0; e < 8; ++e) sc[n][e] = 0.f;
    #pragma unroll
    for (int s = 0; s < 2; ++s) {
      #pragma unroll
      for (int nt = 0; nt < 2; ++nt) {
        const v16bf kf = frag_from_f32(
            kh + (size_t)(j * 32 + nt * 16 + lr) * HD_ + s * 32 + kb);
        sc[nt] = __builtin_amdgcn_wmma_f32_16x16x32_bf16(
            false, qa[s], false, kf, (short)0, sc[nt], false, false);
      }
    }
    #pragma unroll
    for (int e = 0; e < 8; ++e) {              // spill C-layout tile to LDS
      Sb[w][hi * 8 + e][lr]      = sc[0][e];
      Sb[w][hi * 8 + e][16 + lr] = sc[1][e];
    }
    lds_fence();

    // lane owns row lr, columns hi*16 .. hi*16+15
    float sv[16], mx = -1e30f;
    #pragma unroll
    for (int c = 0; c < 16; ++c) {
      const int key = j * 32 + hi * 16 + c;
      float x = Sb[w][lr][hi * 16 + c] * 0.125f;       // 1/sqrt(64)
      if (causal && key > qglob) x = -1e30f;
      sv[c] = x;
      mx = fmaxf(mx, x);
    }
    mx = fmaxf(mx, __shfl_xor(mx, 16, 32));
    const float mnew  = fmaxf(m, mx);
    const float alpha = __expf(m - mnew);
    float ps = 0.f;
    #pragma unroll
    for (int c = 0; c < 16; ++c) {
      const float p = __expf(sv[c] - mnew);
      ps += p;
      Pb[w][lr][hi * 16 + c] = (__bf16)p;
    }
    ps += __shfl_xor(ps, 16, 32);
    l = l * alpha + ps;
    m = mnew;
    if (hi == 0) Al[w][lr] = alpha;
    lds_fence();

    float av[8];
    #pragma unroll
    for (int e = 0; e < 8; ++e) av[e] = Al[w][hi * 8 + e];
    #pragma unroll
    for (int n = 0; n < 4; ++n)
      #pragma unroll
      for (int e = 0; e < 8; ++e) o[n][e] *= av[e];

    // P (16x32) as A fragment: aligned contiguous runs -> ds_load_b128 pair
    const v16bf pa = load_frag_lds(&Pb[w][lr][kb]);
    #pragma unroll
    for (int nt = 0; nt < 4; ++nt) {           // O += P @ V (4 N-tiles of 16)
      Frag vf;                                 // key-major V => strided gather
      #pragma unroll
      for (int e = 0; e < 8; ++e) {
        vf.h[e]     = (__bf16)vh[(size_t)(j * 32 + kb + e) * HD_ + nt * 16 + lr];
        vf.h[e + 8] = (__bf16)vh[(size_t)(j * 32 + 16 + kb + e) * HD_ + nt * 16 + lr];
      }
      o[nt] = __builtin_amdgcn_wmma_f32_16x16x32_bf16(
          false, pa, false, vf.v, (short)0, o[nt], false, false);
    }
  }

  if (hi == 0) Ll[w][lr] = l;
  lds_fence();
  #pragma unroll
  for (int nt = 0; nt < 4; ++nt) {
    #pragma unroll
    for (int e = 0; e < 8; ++e) {
      const int srow = qt * 16 + hi * 8 + e;
      const float linv = 1.f / Ll[w][hi * 8 + e];
      // transpose(0,2,1,3).reshape: out[b, s, h*64+hd]
      out[((size_t)b * S_ + srow) * DM_ + h * HD_ + nt * 16 + lr] = o[nt][e] * linv;
    }
  }
}

// ---------------------------------------------------------------------------
// out = LayerNorm(x + h) * g + b ; one 256-thread block per row of 1024
// ---------------------------------------------------------------------------
__global__ __launch_bounds__(256) void addln_kernel(
    const float* __restrict__ x, const float* __restrict__ hh,
    const float* __restrict__ g, const float* __restrict__ bta,
    float* __restrict__ out)
{
  __shared__ float red[256];
  const int row = blockIdx.x, tid = threadIdx.x;
  const float* xr = x  + (size_t)row * DM_;
  const float* hr = hh + (size_t)row * DM_;
  float vv[4]; float s = 0.f;
  #pragma unroll
  for (int i = 0; i < 4; ++i) { vv[i] = xr[tid + 256 * i] + hr[tid + 256 * i]; s += vv[i]; }
  red[tid] = s; __syncthreads();
  for (int o2 = 128; o2 > 0; o2 >>= 1) { if (tid < o2) red[tid] += red[tid + o2]; __syncthreads(); }
  const float mu = red[0] * (1.f / DM_);
  __syncthreads();
  s = 0.f;
  #pragma unroll
  for (int i = 0; i < 4; ++i) { const float d = vv[i] - mu; s += d * d; }
  red[tid] = s; __syncthreads();
  for (int o2 = 128; o2 > 0; o2 >>= 1) { if (tid < o2) red[tid] += red[tid + o2]; __syncthreads(); }
  const float rstd = rsqrtf(red[0] * (1.f / DM_) + 1e-5f);
  #pragma unroll
  for (int i = 0; i < 4; ++i) {
    const int c = tid + 256 * i;
    out[(size_t)row * DM_ + c] = (vv[i] - mu) * rstd * g[c] + bta[c];
  }
}

// ---------------------------------------------------------------------------
extern "C" void kernel_launch(void* const* d_in, const int* in_sizes, int n_in,
                              void* d_out, int out_size, void* d_ws, size_t ws_size,
                              hipStream_t stream)
{
  (void)in_sizes; (void)n_in; (void)out_size; (void)ws_size;
  const float* x    = (const float*)d_in[0];
  const float* enc  = (const float*)d_in[1];
  const float* s_wq = (const float*)d_in[2];  const float* s_bq = (const float*)d_in[3];
  const float* s_wk = (const float*)d_in[4];  const float* s_bk = (const float*)d_in[5];
  const float* s_wv = (const float*)d_in[6];  const float* s_bv = (const float*)d_in[7];
  const float* s_wo = (const float*)d_in[8];  const float* s_bo = (const float*)d_in[9];
  const float* c_wq = (const float*)d_in[10]; const float* c_bq = (const float*)d_in[11];
  const float* c_wk = (const float*)d_in[12]; const float* c_bk = (const float*)d_in[13];
  const float* c_wv = (const float*)d_in[14]; const float* c_bv = (const float*)d_in[15];
  const float* c_wo = (const float*)d_in[16]; const float* c_bo = (const float*)d_in[17];
  const float* f_w1 = (const float*)d_in[18]; const float* f_b1 = (const float*)d_in[19];
  const float* f_w2 = (const float*)d_in[20]; const float* f_b2 = (const float*)d_in[21];
  const float* ln1g = (const float*)d_in[22]; const float* ln1b = (const float*)d_in[23];
  const float* ln2g = (const float*)d_in[24]; const float* ln2b = (const float*)d_in[25];
  const float* ln3g = (const float*)d_in[26]; const float* ln3b = (const float*)d_in[27];
  float* out = (float*)d_out;

  const size_t R = (size_t)B_ * S_;            // 4096 token rows
  float* ws  = (float*)d_ws;
  float* Q   = ws;
  float* K   = Q   + R * DM_;
  float* V   = K   + R * DM_;
  float* ATT = V   + R * DM_;
  float* Hh  = ATT + R * DM_;
  float* Y1  = Hh  + R * DM_;
  float* Y2  = Y1  + R * DM_;
  float* FF  = Y2  + R * DM_;                  // R * DFF_ floats

  const dim3 blk(256);
  const dim3 gP(DM_ / 64, (unsigned)(R / 128));
  const dim3 gF1(DFF_ / 64, (unsigned)(R / 128));
  const dim3 aGrid((B_ * H_ * (S_ / 16)) / 4);
  const dim3 aBlk(128);
  const dim3 lnGrid((unsigned)R);

  // --- masked self-attention ---
  gemm_bf16_kernel<<<gP, blk, 0, stream>>>(x, s_wq, s_bq, Q, (int)R, DM_, DM_, 0);
  gemm_bf16_kernel<<<gP, blk, 0, stream>>>(x, s_wk, s_bk, K, (int)R, DM_, DM_, 0);
  gemm_bf16_kernel<<<gP, blk, 0, stream>>>(x, s_wv, s_bv, V, (int)R, DM_, DM_, 0);
  attn_kernel<<<aGrid, aBlk, 0, stream>>>(Q, K, V, ATT, 1);
  gemm_bf16_kernel<<<gP, blk, 0, stream>>>(ATT, s_wo, s_bo, Hh, (int)R, DM_, DM_, 0);
  addln_kernel<<<lnGrid, blk, 0, stream>>>(x, Hh, ln1g, ln1b, Y1);

  // --- cross-attention ---
  gemm_bf16_kernel<<<gP, blk, 0, stream>>>(Y1, c_wq, c_bq, Q, (int)R, DM_, DM_, 0);
  gemm_bf16_kernel<<<gP, blk, 0, stream>>>(enc, c_wk, c_bk, K, (int)R, DM_, DM_, 0);
  gemm_bf16_kernel<<<gP, blk, 0, stream>>>(enc, c_wv, c_bv, V, (int)R, DM_, DM_, 0);
  attn_kernel<<<aGrid, aBlk, 0, stream>>>(Q, K, V, ATT, 0);
  gemm_bf16_kernel<<<gP, blk, 0, stream>>>(ATT, c_wo, c_bo, Hh, (int)R, DM_, DM_, 0);
  addln_kernel<<<lnGrid, blk, 0, stream>>>(Y1, Hh, ln2g, ln2b, Y2);

  // --- FFN ---
  gemm_bf16_kernel<<<gF1, blk, 0, stream>>>(Y2, f_w1, f_b1, FF, (int)R, DM_, DFF_, 1);
  gemm_bf16_kernel<<<gP, blk, 0, stream>>>(FF, f_w2, f_b2, Hh, (int)R, DFF_, DM_, 0);
  addln_kernel<<<lnGrid, blk, 0, stream>>>(Y2, Hh, ln3g, ln3b, out);
}
